// TDAN_align_15582141350129
// MI455X (gfx1250) — compile-verified, matrix operand order checked by reading, not run on hardware
//
#include <hip/hip_runtime.h>
#include <hip/hip_bf16.h>

typedef __bf16 bf16_t;
typedef bf16_t bf16x8  __attribute__((ext_vector_type(8)));
typedef bf16_t bf16x16 __attribute__((ext_vector_type(16)));
typedef float  f32x8   __attribute__((ext_vector_type(8)));

#define NB   4
#define HH   128
#define WW   128
#define HWSZ (HH * WW)
#define NPIX (NB * HWSZ)

__device__ __forceinline__ bf16x8 bzero8() {
    bf16x8 z;
#pragma unroll
    for (int i = 0; i < 8; i++) z[i] = (bf16_t)0.0f;
    return z;
}
__device__ __forceinline__ bf16x16 bcat(bf16x8 lo, bf16x8 hi) {
    return __builtin_shufflevector(lo, hi, 0, 1, 2, 3, 4, 5, 6, 7,
                                   8, 9, 10, 11, 12, 13, 14, 15);
}

// ---------------------------------------------------------------------------
// NCHW f32 -> NHWC bf16 (optionally into a channel slice of a wider buffer)
// ---------------------------------------------------------------------------
__global__ void k_to_nhwc_bf16(const float* __restrict__ src, bf16_t* __restrict__ dst,
                               int Csrc, int Cdst, int coff) {
    int idx = blockIdx.x * blockDim.x + threadIdx.x;
    int total = NB * Csrc * HWSZ;
    if (idx >= total) return;
    int pix = idx % HWSZ;
    int c   = (idx / HWSZ) % Csrc;
    int b   = idx / (HWSZ * Csrc);
    dst[(size_t)(b * HWSZ + pix) * Cdst + coff + c] = (bf16_t)src[idx];
}

// ---------------------------------------------------------------------------
// Pack OIHW f32 weights into WMMA B-fragment order, bf16.
// 16-bit B 32x16 layout: lane<16 -> N=lane, K {0..7,16..23};
// lane>=16 -> N=lane-16, K {8..15,24..31}.  kk = tap*CIN + cin.
// Linear: [ntile][kstep][lane][e], 16 contiguous bf16 per lane.
// ---------------------------------------------------------------------------
template <int CIN, int COUT>
__global__ void k_pack_w(const float* __restrict__ w, bf16_t* __restrict__ wf) {
    const int KSTEPS = CIN * 9 / 32;
    int idx = blockIdx.x * blockDim.x + threadIdx.x;
    int total = COUT * CIN * 9;
    if (idx >= total) return;
    int e    = idx & 15;
    int lane = (idx >> 4) & 31;
    int s    = (idx >> 9) % KSTEPS;
    int nt   = idx / (512 * KSTEPS);
    int kk   = s * 32 + ((lane >> 4) << 3) + (e < 8 ? e : e + 8);
    int ch   = nt * 16 + (lane & 15);
    int tap  = kk / CIN;
    int c    = kk % CIN;
    wf[idx]  = (bf16_t)w[((size_t)ch * CIN + c) * 9 + tap];
}

// ---------------------------------------------------------------------------
// 3x3 conv (pad=1) as implicit GEMM on v_wmma_f32_16x16x32_bf16.
// WG: COUT/16 waves x 4 M-tiles (64 consecutive pixels of one row).
// Input halo tile staged in LDS chunk-major [c/8][r][x][8ch] so inner-loop
// A reads are branch-free, conflict-free ds_load_b128.
// OUTMODE 0: bf16 NHWC; 1: f32 NHWC.
// ---------------------------------------------------------------------------
template <int CIN, int COUT, int OUTMODE>
__global__ __launch_bounds__((COUT / 16) * 32)
void k_conv3x3_wmma(const bf16_t* __restrict__ in, const bf16_t* __restrict__ wf,
                    const float* __restrict__ bias, void* __restrict__ outv) {
    const int KSTEPS = CIN * 9 / 32;
    const int MT   = 4;
    const int TW   = 66;          // 64 pixels + halo
    const int CH8  = CIN / 8;     // 16B chunks along channels
    const int NTHR = (COUT / 16) * 32;
    // layout: tile[chunk][r*TW + x] of 8-channel groups
    __shared__ __align__(16) bf16_t tile[CH8 * 3 * TW * 8];

    int tid  = threadIdx.x;
    int p0   = blockIdx.x * 64;
    int b    = p0 / HWSZ;
    int row  = (p0 % HWSZ) / WW;
    int x0   = p0 % WW;

    // ---- cooperative halo staging (zero padding materialized here) ----
    const int NCHUNK = CH8 * 3 * TW;
    for (int ch = tid; ch < NCHUNK; ch += NTHR) {
        int xx = ch % TW;
        int rr = (ch / TW) % 3;
        int cc = ch / (3 * TW);
        int iy = row + rr - 1;
        int ix = x0 + xx - 1;
        bf16x8 v = bzero8();
        if (iy >= 0 && iy < HH && ix >= 0 && ix < WW)
            v = *(const bf16x8*)(in + (size_t)(b * HWSZ + iy * WW + ix) * CIN + cc * 8);
        *(bf16x8*)&tile[(size_t)ch * 8] = v;
    }
    __syncthreads();

    int wave = tid >> 5;
    int lane = tid & 31;
    int m    = lane & 15;
    int hi   = lane >> 4;

    f32x8 acc[MT];
#pragma unroll
    for (int t = 0; t < MT; t++)
#pragma unroll
        for (int r = 0; r < 8; r++) acc[t][r] = 0.0f;

    const bf16_t* wfw = wf + (size_t)wave * KSTEPS * 512;

    for (int s = 0; s < KSTEPS; s++) {
        const bf16_t* bp = wfw + ((size_t)s * 32 + lane) * 16;
        bf16x16 bfrag = bcat(*(const bf16x8*)bp, *(const bf16x8*)(bp + 8));

        int tap = (s * 32) / CIN;               // fixed within K-step
        int c0  = (s * 32) % CIN + hi * 8;      // multiple of 8
        int ty  = tap / 3, tx = tap % 3;        // 0..2 into halo tile
        int ck0 = c0 >> 3;
        // chunk-major LDS address (in 8-elem units)
        int base = (ck0 * 3 + ty) * TW + tx;    // + pixel x gives chunk index

#pragma unroll
        for (int t = 0; t < MT; t++) {
            int px = t * 16 + m;
            const bf16_t* ap = &tile[(size_t)(base + px) * 8];
            // second 16B sits 2 chunks (16 channels) further
            const bf16_t* ap2 = ap + (size_t)2 * 3 * TW * 8;
            bf16x16 a = bcat(*(const bf16x8*)ap, *(const bf16x8*)ap2);
            acc[t] = __builtin_amdgcn_wmma_f32_16x16x32_bf16(
                false, a, false, bfrag, (short)0, acc[t], false, false);
        }
    }

    int ch = wave * 16 + m;
    float bv = bias[ch];
#pragma unroll
    for (int t = 0; t < MT; t++) {
#pragma unroll
        for (int r = 0; r < 8; r++) {
            int p = p0 + t * 16 + r + hi * 8;   // C/D: M = r + 8*hi, N = m
            float v = acc[t][r] + bv;
            if (OUTMODE == 0) ((bf16_t*)outv)[(size_t)p * COUT + ch] = (bf16_t)v;
            else              ((float*)outv)[(size_t)p * COUT + ch]  = v;
        }
    }
}

// ---------------------------------------------------------------------------
// Deformable conv (DG=8 groups, K=9 taps, Cin=Cout=64).
// Phase 1: 128 threads bilinearly sample a 16-pixel x 576-K im2col tile into
// LDS, chunk-major cols[kk/8][pixel][8ch] (conflict-free phase-2 reads).
// Phase 2: 4 waves x one 16x16 N-tile each over 18 WMMA K-steps.
// FINAL=1 writes f32 NCHW (network output), else bf16 NHWC.
// ---------------------------------------------------------------------------
template <int FINAL>
__global__ __launch_bounds__(128)
void k_deform_wmma(const bf16_t* __restrict__ img, const float* __restrict__ off,
                   const bf16_t* __restrict__ wf, const float* __restrict__ bias,
                   void* __restrict__ outv) {
    const int CIN = 64, COUT = 64, KSTEPS = 18, CG = 8;
    // cols[chunk=kk/8][pixel] of 8-channel groups: 72*16*16B = 18.4 KB
    __shared__ __align__(16) bf16_t cols[72 * 16 * 8];

    int tid = threadIdx.x;
    int p0  = blockIdx.x * 16;
    int b   = p0 / HWSZ;
    int row = (p0 % HWSZ) / WW;
    int x0  = p0 % WW;

    // ---- phase 1: sampling (16 px * 9 taps * 8 groups = 1152 jobs) ----
    for (int it = 0; it < 9; it++) {
        int job  = it * 128 + tid;
        int m    = job / 72;
        int rest = job % 72;
        int tap  = rest / 8;
        int dg   = rest % 8;
        int x    = x0 + m;

        const float* op = off + (size_t)(p0 + m) * 144 + dg * 18 + tap * 2;
        float dy = op[0], dx = op[1];
        float py = dy + (float)row + (float)(tap / 3 - 1);
        float px = dx + (float)x   + (float)(tap % 3 - 1);
        float fy = floorf(py), fx = floorf(px);
        float wy = py - fy, wx = px - fx;
        int iy0 = (int)fy, ix0 = (int)fx;

        float v[4][CG];
        int yy[2] = {iy0, iy0 + 1};
        int xx[2] = {ix0, ix0 + 1};
#pragma unroll
        for (int cy = 0; cy < 2; cy++)
#pragma unroll
            for (int cx = 0; cx < 2; cx++) {
                bool ok = (yy[cy] >= 0) && (yy[cy] < HH) && (xx[cx] >= 0) && (xx[cx] < WW);
                int yc = min(max(yy[cy], 0), HH - 1);
                int xc = min(max(xx[cx], 0), WW - 1);
                const bf16_t* gp = img + (size_t)(b * HWSZ + yc * WW + xc) * CIN + dg * CG;
#pragma unroll
                for (int i = 0; i < CG; i++)
                    v[cy * 2 + cx][i] = ok ? (float)gp[i] : 0.0f;
            }
        float w00 = (1.f - wy) * (1.f - wx), w01 = (1.f - wy) * wx;
        float w10 = wy * (1.f - wx),         w11 = wy * wx;
        int chunk = tap * 8 + dg;              // kk/8
        bf16_t* cp = &cols[(size_t)(chunk * 16 + m) * 8];
        bf16x8 r8;
#pragma unroll
        for (int i = 0; i < CG; i++)
            r8[i] = (bf16_t)(v[0][i] * w00 + v[1][i] * w01 + v[2][i] * w10 + v[3][i] * w11);
        *(bf16x8*)cp = r8;
    }
    __syncthreads();

    // ---- phase 2: WMMA GEMM over K = 576 ----
    int wave = tid >> 5, lane = tid & 31;
    int m  = lane & 15;
    int hi = lane >> 4;

    f32x8 acc;
#pragma unroll
    for (int r = 0; r < 8; r++) acc[r] = 0.0f;

    const bf16_t* wfw = wf + (size_t)wave * KSTEPS * 512;
    for (int s = 0; s < KSTEPS; s++) {
        const bf16_t* bp = wfw + ((size_t)s * 32 + lane) * 16;
        bf16x16 bfrag = bcat(*(const bf16x8*)bp, *(const bf16x8*)(bp + 8));

        int ck = (s * 32 + hi * 8) >> 3;       // base chunk for this lane
        const bf16_t* ap  = &cols[(size_t)(ck * 16 + m) * 8];
        const bf16_t* ap2 = &cols[(size_t)((ck + 2) * 16 + m) * 8];  // +16 channels
        bf16x16 a = bcat(*(const bf16x8*)ap, *(const bf16x8*)ap2);

        acc = __builtin_amdgcn_wmma_f32_16x16x32_bf16(
            false, a, false, bfrag, (short)0, acc, false, false);
    }

    int ch = wave * 16 + m;
    float bv = bias[ch];
#pragma unroll
    for (int r = 0; r < 8; r++) {
        int p = p0 + r + hi * 8;
        float v = acc[r] + bv;
        if (FINAL) {
            int bb = p / HWSZ, pix = p % HWSZ;
            ((float*)outv)[((size_t)bb * COUT + ch) * HWSZ + pix] = v;
        } else {
            ((bf16_t*)outv)[(size_t)p * COUT + ch] = (bf16_t)v;
        }
    }
}

// ---------------------------------------------------------------------------
// Host-side pipeline
// ---------------------------------------------------------------------------
extern "C" void kernel_launch(void* const* d_in, const int* in_sizes, int n_in,
                              void* d_out, int out_size, void* d_ws, size_t ws_size,
                              hipStream_t stream) {
    (void)in_sizes; (void)n_in; (void)out_size; (void)ws_size;

    const float* neibor = (const float*)d_in[0];
    const float* target = (const float*)d_in[1];
    const float* cr_w   = (const float*)d_in[2];
    const float* cr_b   = (const float*)d_in[3];
    const float* off_w[4] = {(const float*)d_in[4],  (const float*)d_in[8],
                             (const float*)d_in[12], (const float*)d_in[16]};
    const float* off_b[4] = {(const float*)d_in[5],  (const float*)d_in[9],
                             (const float*)d_in[13], (const float*)d_in[17]};
    const float* dw[4]    = {(const float*)d_in[6],  (const float*)d_in[10],
                             (const float*)d_in[14], (const float*)d_in[18]};
    const float* db[4]    = {(const float*)d_in[7],  (const float*)d_in[11],
                             (const float*)d_in[15], (const float*)d_in[19]};

    char* ws = (char*)d_ws;
    size_t cur = 0;
    auto alloc = [&](size_t bytes) -> void* {
        void* p = ws + cur;
        cur = (cur + bytes + 255) & ~(size_t)255;
        return p;
    };
    bf16_t* concat128 = (bf16_t*)alloc((size_t)NPIX * 128 * 2);
    bf16_t* supp      = (bf16_t*)alloc((size_t)NPIX * 64 * 2);
    bf16_t* feaA      = (bf16_t*)alloc((size_t)NPIX * 64 * 2);
    bf16_t* feaB      = (bf16_t*)alloc((size_t)NPIX * 64 * 2);
    float*  offbuf    = (float*)alloc((size_t)NPIX * 144 * 4);
    bf16_t* wf_cr     = (bf16_t*)alloc((size_t)128 * 9 * 64 * 2);
    bf16_t* wf_off[4];
    bf16_t* wf_d[4];
    for (int i = 0; i < 4; i++) wf_off[i] = (bf16_t*)alloc((size_t)64 * 9 * 144 * 2);
    for (int i = 0; i < 4; i++) wf_d[i]   = (bf16_t*)alloc((size_t)64 * 9 * 64 * 2);

    const int T = 256;
    int nconv = NPIX * 64;
    k_to_nhwc_bf16<<<(nconv + T - 1) / T, T, 0, stream>>>(neibor, concat128, 64, 128, 0);
    k_to_nhwc_bf16<<<(nconv + T - 1) / T, T, 0, stream>>>(target, concat128, 64, 128, 64);
    k_to_nhwc_bf16<<<(nconv + T - 1) / T, T, 0, stream>>>(neibor, supp, 64, 64, 0);

    k_pack_w<128, 64><<<(128 * 64 * 9 + T - 1) / T, T, 0, stream>>>(cr_w, wf_cr);
    for (int i = 0; i < 4; i++)
        k_pack_w<64, 144><<<(64 * 144 * 9 + T - 1) / T, T, 0, stream>>>(off_w[i], wf_off[i]);
    for (int i = 0; i < 4; i++)
        k_pack_w<64, 64><<<(64 * 64 * 9 + T - 1) / T, T, 0, stream>>>(dw[i], wf_d[i]);

    // fea0 = conv_cr(concat)
    k_conv3x3_wmma<128, 64, 0><<<NPIX / 64, 128, 0, stream>>>(concat128, wf_cr, cr_b, feaA);
    // o1 / deform1
    k_conv3x3_wmma<64, 144, 1><<<NPIX / 64, 288, 0, stream>>>(feaA, wf_off[0], off_b[0], offbuf);
    k_deform_wmma<0><<<NPIX / 16, 128, 0, stream>>>(feaA, offbuf, wf_d[0], db[0], feaB);
    // o2 / deform2
    k_conv3x3_wmma<64, 144, 1><<<NPIX / 64, 288, 0, stream>>>(feaB, wf_off[1], off_b[1], offbuf);
    k_deform_wmma<0><<<NPIX / 16, 128, 0, stream>>>(feaB, offbuf, wf_d[1], db[1], feaA);
    // o3 / deform3 (samples supp!)
    k_conv3x3_wmma<64, 144, 1><<<NPIX / 64, 288, 0, stream>>>(feaA, wf_off[2], off_b[2], offbuf);
    k_deform_wmma<0><<<NPIX / 16, 128, 0, stream>>>(supp, offbuf, wf_d[2], db[2], feaB);
    // o4 / deform4 -> final f32 NCHW
    k_conv3x3_wmma<64, 144, 1><<<NPIX / 64, 288, 0, stream>>>(feaB, wf_off[3], off_b[3], offbuf);
    k_deform_wmma<1><<<NPIX / 16, 128, 0, stream>>>(feaB, offbuf, wf_d[3], db[3], d_out);
}
